// SupConLoss_13091060319093
// MI455X (gfx1250) — matrix operand order repaired and checked by reading, not compiled, not run on hardware
//
#include <hip/hip_runtime.h>

typedef __attribute__((ext_vector_type(16))) _Float16 v16h;
typedef __attribute__((ext_vector_type(8)))  _Float16 half8;
typedef __attribute__((ext_vector_type(8)))  float    v8f;

#define TILE_M   256     // block rows
#define TILE_N   128     // block cols
#define TILE_K   32
#define LDSS     40      // halfs per LDS row (80 B, multiple of 16 B)
#define COLS_PER_BLK 4   // column tiles swept per block
#define INV_T    2.0f    // 1 / TEMPERATURE (0.5)

// CDNA5 async global->LDS path (ASYNCcnt), guarded so a missing builtin
// falls back to the proven synchronous staging path.
#if defined(__gfx1250__) && __has_builtin(__builtin_amdgcn_global_load_async_to_lds_b128)
#define USE_ASYNC_LDS 1
#else
#define USE_ASYNC_LDS 0
#endif

#if USE_ASYNC_LDS
// Probe-derived signature: parameters are generic pointers to int vector_size(16).
typedef int async_v4i __attribute__((vector_size(16)));
#define ASYNC_B128(g, l, imm_off) \
    __builtin_amdgcn_global_load_async_to_lds_b128( \
        (async_v4i*)(g), (async_v4i*)(l), (imm_off), 0)
__device__ __forceinline__ void wait_async0() {
#if __has_builtin(__builtin_amdgcn_s_wait_asynccnt)
    __builtin_amdgcn_s_wait_asynccnt(0);
#else
    asm volatile("s_wait_asynccnt 0x0" ::: "memory");
#endif
}
#endif

// ---------------- zero the denominator accumulators ----------------
__global__ void ntxent_init(float* __restrict__ denom, int n) {
    int i = blockIdx.x * blockDim.x + threadIdx.x;
    if (i < n) denom[i] = 0.0f;
}

// -------- row L2-normalize; emit Z (f16) and inverse norms (f32) --------
__global__ __launch_bounds__(256) void ntxent_normalize(
    const float* __restrict__ emb_i, const float* __restrict__ emb_j,
    _Float16* __restrict__ Z, float* __restrict__ invn, int B, int D) {
    int row = blockIdx.x;                 // 0 .. 2B-1
    int tid = threadIdx.x;                // 256 threads, D=1024 -> float4 each
    const float* src = (row < B) ? (emb_i + (size_t)row * D)
                                 : (emb_j + (size_t)(row - B) * D);
    float4 x = ((const float4*)src)[tid];
    float ss = x.x * x.x + x.y * x.y + x.z * x.z + x.w * x.w;
    __shared__ float sh[256];
    sh[tid] = ss; __syncthreads();
    for (int s = 128; s > 0; s >>= 1) {
        if (tid < s) sh[tid] += sh[tid + s];
        __syncthreads();
    }
    float inv = 1.0f / fmaxf(sqrtf(sh[0]), 1e-12f);
    _Float16* zr = Z + (size_t)row * D + tid * 4;
    zr[0] = (_Float16)(x.x * inv); zr[1] = (_Float16)(x.y * inv);
    zr[2] = (_Float16)(x.z * inv); zr[3] = (_Float16)(x.w * inv);
    if (tid == 0) invn[row] = inv;
}

// -------- positives in full fp32 from raw inputs: pos[n] = <z_i[n], z_j[n]> --------
__global__ __launch_bounds__(256) void ntxent_positives(
    const float* __restrict__ emb_i, const float* __restrict__ emb_j,
    const float* __restrict__ invn, float* __restrict__ pos, int B, int D) {
    int n = blockIdx.x, tid = threadIdx.x;
    float4 a = ((const float4*)(emb_i + (size_t)n * D))[tid];
    float4 b = ((const float4*)(emb_j + (size_t)n * D))[tid];
    float d = a.x * b.x + a.y * b.y + a.z * b.z + a.w * b.w;
    __shared__ float sh[256];
    sh[tid] = d; __syncthreads();
    for (int s = 128; s > 0; s >>= 1) {
        if (tid < s) sh[tid] += sh[tid + s];
        __syncthreads();
    }
    if (tid == 0) pos[n] = sh[0] * invn[n] * invn[B + n];
}

// -------- fused sim = Z*Z^T (WMMA f16->f32) + rowwise sum of exp(sim/T), diag excluded --------
// Block tile 256x128, 8 waves in 4(M) x 2(N), each wave 64x64 -> 16 WMMA accumulators.
// Async path: double-buffered LDS K-panels fed by global_load_async_to_lds_b128,
// one barrier per K-step, next panel in flight while WMMAs consume current.
__global__ __launch_bounds__(256) void ntxent_gemm_exp(
    const _Float16* __restrict__ Z, float* __restrict__ denom, int N, int D) {
#if USE_ASYNC_LDS
    __shared__ _Float16 As[2][TILE_M * LDSS];   // 2 x (256 x 32 halfs, padded)
    __shared__ _Float16 Bs[2][TILE_N * LDSS];   // 2 x (128 x 32 halfs, padded)
#else
    __shared__ _Float16 As[TILE_M * LDSS];
    __shared__ _Float16 Bs[TILE_N * LDSS];
#endif

    const int rowBase = blockIdx.x * TILE_M;
    const int tid  = threadIdx.x;
    const int lane = tid & 31;
    const int wave = tid >> 5;          // 8 waves
    const int waveM = wave & 3;         // 4 waves down  -> 64 rows each
    const int waveN = wave >> 2;        // 2 waves across -> 64 cols each
    const int lr = lane & 15;           // lane-in-half
    const int lh = lane >> 4;           // half id

    // global-load coords: As -> 1 full 32-half row per thread; Bs -> half row per thread
    const int bRow   = tid >> 1;
    const int bChunk = (tid & 1) * 16;

    const int nk = D / TILE_K;

    float rsum[4][8];
    #pragma unroll
    for (int mt = 0; mt < 4; ++mt)
        #pragma unroll
        for (int v = 0; v < 8; ++v) rsum[mt][v] = 0.0f;

    const v8f vzero = {0.f, 0.f, 0.f, 0.f, 0.f, 0.f, 0.f, 0.f};

    for (int ct = 0; ct < COLS_PER_BLK; ++ct) {
        const int colTile = blockIdx.y * (COLS_PER_BLK * TILE_N) + ct * TILE_N;
        v8f acc[4][4];
        #pragma unroll
        for (int mt = 0; mt < 4; ++mt)
            #pragma unroll
            for (int nt = 0; nt < 4; ++nt) acc[mt][nt] = vzero;

#if USE_ASYNC_LDS
        // prologue: async-fill buffer 0 with the first K-panel (6 x b128 per lane,
        // same INST_OFFSET applies to both global and LDS addresses per the ISA)
        {
            const _Float16* gA = Z + (size_t)(rowBase + tid) * D;
            const _Float16* gB = Z + (size_t)(colTile + bRow) * D + bChunk;
            _Float16* lA = &As[0][tid * LDSS];
            _Float16* lB = &Bs[0][bRow * LDSS + bChunk];
            ASYNC_B128(gA, lA, 0);  ASYNC_B128(gA, lA, 16);
            ASYNC_B128(gA, lA, 32); ASYNC_B128(gA, lA, 48);
            ASYNC_B128(gB, lB, 0);  ASYNC_B128(gB, lB, 16);
        }
#endif

        for (int ki = 0; ki < nk; ++ki) {
            const int k0 = ki * TILE_K;
#if USE_ASYNC_LDS
            const int cur = ki & 1;
            wait_async0();          // own loads for buf[cur] have landed in LDS
            __syncthreads();        // everyone's landed; prior reads of buf[cur^1] done
            if (ki + 1 < nk) {      // stream next panel into buf[cur^1] during compute
                const _Float16* gA = Z + (size_t)(rowBase + tid) * D + k0 + TILE_K;
                const _Float16* gB = Z + (size_t)(colTile + bRow) * D + k0 + TILE_K + bChunk;
                _Float16* lA = &As[cur ^ 1][tid * LDSS];
                _Float16* lB = &Bs[cur ^ 1][bRow * LDSS + bChunk];
                ASYNC_B128(gA, lA, 0);  ASYNC_B128(gA, lA, 16);
                ASYNC_B128(gA, lA, 32); ASYNC_B128(gA, lA, 48);
                ASYNC_B128(gB, lB, 0);  ASYNC_B128(gB, lB, 16);
            }
            const _Float16* Acur = As[cur];
            const _Float16* Bcur = Bs[cur];
#else
            __syncthreads();   // previous iteration's fragment reads done
            const _Float16* gA = Z + (size_t)(rowBase + tid) * D + k0;
            const _Float16* gB = Z + (size_t)(colTile + bRow) * D + k0 + bChunk;
            #pragma unroll
            for (int c = 0; c < 4; ++c)
                *(half8*)&As[tid * LDSS + c * 8] = ((const half8*)gA)[c];
            *(half8*)&Bs[bRow * LDSS + bChunk]     = *(const half8*)(gB);
            *(half8*)&Bs[bRow * LDSS + bChunk + 8] = *(const half8*)(gB + 8);
            __syncthreads();
            if (k0 + TILE_K < D) {
                __builtin_prefetch((const void*)(gA + TILE_K), 0, 1);
                __builtin_prefetch((const void*)(gB + TILE_K), 0, 1);
            }
            const _Float16* Acur = As;
            const _Float16* Bcur = Bs;
#endif

            // A fragments: lane(r=lr,h=lh) holds K = {8h..8h+7} U {16+8h..+7} (ISA 16-bit A 16x32)
            v16h afrag[4];
            #pragma unroll
            for (int mt = 0; mt < 4; ++mt) {
                const _Float16* base = &Acur[(waveM * 64 + mt * 16 + lr) * LDSS];
                half8 lo = *(const half8*)(base + lh * 8);
                half8 hi = *(const half8*)(base + 16 + lh * 8);
                afrag[mt] = __builtin_shufflevector(lo, hi,
                    0,1,2,3,4,5,6,7, 8,9,10,11,12,13,14,15);
            }
            // B fragments: lane(n=lr,h=lh) holds K = {8h..8h+7} U {16+8h..+7} of column row n
            #pragma unroll
            for (int nt = 0; nt < 4; ++nt) {
                const _Float16* base = &Bcur[(waveN * 64 + nt * 16 + lr) * LDSS];
                half8 lo = *(const half8*)(base + lh * 8);
                half8 hi = *(const half8*)(base + 16 + lh * 8);
                v16h bfrag = __builtin_shufflevector(lo, hi,
                    0,1,2,3,4,5,6,7, 8,9,10,11,12,13,14,15);
                #pragma unroll
                for (int mt = 0; mt < 4; ++mt)
                    acc[mt][nt] = __builtin_amdgcn_wmma_f32_16x16x32_f16(
                        false, afrag[mt], false, bfrag,
                        (short)0, acc[mt][nt], false, false);
            }
        }

        // epilogue: exp(sim/T) with diagonal masked, accumulate per-row partials
        #pragma unroll
        for (int mt = 0; mt < 4; ++mt) {
            const int growBase = rowBase + waveM * 64 + mt * 16 + 8 * lh;
            #pragma unroll
            for (int nt = 0; nt < 4; ++nt) {
                const int gcol = colTile + waveN * 64 + nt * 16 + lr;
                #pragma unroll
                for (int v = 0; v < 8; ++v) {
                    float e = __expf(acc[mt][nt][v] * INV_T);
                    if (growBase + v == gcol) e = 0.0f;
                    rsum[mt][v] += e;
                }
            }
        }
    }

    // cross-lane reduce within each 16-lane half, then fp32 atomics
    #pragma unroll
    for (int mt = 0; mt < 4; ++mt) {
        #pragma unroll
        for (int v = 0; v < 8; ++v) {
            float val = rsum[mt][v];
            val += __shfl_xor(val, 1, 32);
            val += __shfl_xor(val, 2, 32);
            val += __shfl_xor(val, 4, 32);
            val += __shfl_xor(val, 8, 32);
            if (lr == 0)
                atomicAdd(&denom[rowBase + waveM * 64 + mt * 16 + 8 * lh + v], val);
        }
    }
}

// -------- final scalar: mean(log(denom)) - mean(pos)/T --------
__global__ __launch_bounds__(256) void ntxent_finalize(
    const float* __restrict__ denom, const float* __restrict__ pos,
    float* __restrict__ out, int N, int B) {
    int tid = threadIdx.x;
    float v = 0.0f;
    for (int i = tid; i < N; i += 256) v += __logf(denom[i]);
    // Sum over 2B positives = 2 * sum_n pos[n]; each divided by T -> 2*INV_T factor
    for (int i = tid; i < B; i += 256) v -= 2.0f * INV_T * pos[i];
    __shared__ float sh[256];
    sh[tid] = v; __syncthreads();
    for (int s = 128; s > 0; s >>= 1) {
        if (tid < s) sh[tid] += sh[tid + s];
        __syncthreads();
    }
    if (tid == 0) out[0] = sh[0] / (float)N;
}

extern "C" void kernel_launch(void* const* d_in, const int* in_sizes, int n_in,
                              void* d_out, int out_size, void* d_ws, size_t ws_size,
                              hipStream_t stream) {
    const float* emb_i = (const float*)d_in[0];
    const float* emb_j = (const float*)d_in[1];
    float* out = (float*)d_out;

    const int D = 1024;
    const int B = in_sizes[0] / D;     // 4096
    const int N = 2 * B;               // 8192

    // workspace layout
    char* ws = (char*)d_ws;
    _Float16* Z   = (_Float16*)ws;
    size_t zbytes = (size_t)N * D * sizeof(_Float16);      // 16 MB
    float* invn   = (float*)(ws + zbytes);
    float* denom  = invn + N;
    float* pos    = denom + N;

    ntxent_init<<<(N + 255) / 256, 256, 0, stream>>>(denom, N);
    ntxent_normalize<<<N, 256, 0, stream>>>(emb_i, emb_j, Z, invn, B, D);
    ntxent_positives<<<B, 256, 0, stream>>>(emb_i, emb_j, invn, pos, B, D);

    dim3 grid(N / TILE_M, N / (COLS_PER_BLK * TILE_N));    // (32, 16) -> 512 blocks
    ntxent_gemm_exp<<<grid, 256, 0, stream>>>(Z, denom, N, D);

    ntxent_finalize<<<1, 256, 0, stream>>>(denom, pos, out, N, B);
}
